// Net_36593121362106
// MI455X (gfx1250) — compile-verified, matrix operand order checked by reading, not run on hardware
//
#include <hip/hip_runtime.h>
#include <hip/hip_bf16.h>

#define NN 100000
#define NE 1600000
#define DD 128

typedef __attribute__((ext_vector_type(16))) _Float16 v16h;
typedef __attribute__((ext_vector_type(8)))  _Float16 v8h;
typedef __attribute__((ext_vector_type(8)))  float    v8f;

// D = A(16x32 f16) x B(32x16 f16) + C(16x16 f32)
static __device__ inline v8f wmma16(v16h a, v16h b, v8f c) {
  return __builtin_amdgcn_wmma_f32_16x16x32_f16(
      /*neg_a=*/false, a, /*neg_b=*/false, b,
      /*c_mod=*/(short)0, c, /*reuse_a=*/false, /*reuse_b=*/false);
}

static __device__ inline v16h combine(v8h lo, v8h hi) {
  v16h r;
#pragma unroll
  for (int i = 0; i < 8; ++i) { r[i] = lo[i]; r[i + 8] = hi[i]; }
  return r;
}

// 16 f32 values of one A-fragment row chunk, loaded up-front into distinct regs
struct F32x16 { float4 a, b, c, d; };

static __device__ inline F32x16 load_row16(const float* __restrict__ row, int k0, int lane) {
  int off = ((lane >> 4) & 1) * 8;
  F32x16 r;
  r.a = *(const float4*)(row + k0 + off);
  r.b = *(const float4*)(row + k0 + off + 4);
  r.c = *(const float4*)(row + k0 + 16 + off);
  r.d = *(const float4*)(row + k0 + 16 + off + 4);
  return r;
}

static __device__ inline v16h cvt_frag(F32x16 f, float s) {
  v16h r;
  r[0]  = (_Float16)(f.a.x * s); r[1]  = (_Float16)(f.a.y * s);
  r[2]  = (_Float16)(f.a.z * s); r[3]  = (_Float16)(f.a.w * s);
  r[4]  = (_Float16)(f.b.x * s); r[5]  = (_Float16)(f.b.y * s);
  r[6]  = (_Float16)(f.b.z * s); r[7]  = (_Float16)(f.b.w * s);
  r[8]  = (_Float16)(f.c.x * s); r[9]  = (_Float16)(f.c.y * s);
  r[10] = (_Float16)(f.c.z * s); r[11] = (_Float16)(f.c.w * s);
  r[12] = (_Float16)(f.d.x * s); r[13] = (_Float16)(f.d.y * s);
  r[14] = (_Float16)(f.d.z * s); r[15] = (_Float16)(f.d.w * s);
  return r;
}

// ---- fragment-ordered LDS weight layout (bank-conflict-free) ----
// idx(kk,c,chunk,lane,j) = ((((kk*8 + c)*2 + chunk)*32 + lane)*8 + j) halves
// A wave32 ds_load_b128 at lane*16B sweeps all 64 banks in min cycles.
static __device__ inline int frag_idx(int kk, int c, int chunk, int lane) {
  return (((kk * 8 + c) * 2 + chunk) * 32 + lane) * 8;
}

static __device__ inline v16h b_frag_f(const _Float16* w, int kk, int c, int lane) {
  const _Float16* base = w + frag_idx(kk, c, 0, lane);
  v8h lo = *(const v8h*)base;
  v8h hi = *(const v8h*)(base + 32 * 8);  // chunk 1
  return combine(lo, hi);
}

// Store 4 consecutive halves of W row n (cols k..k+3, k%4==0) into frag layout.
// B[k][n] = W[n][k]: element (n,k) -> c=n>>4, lane=(km>>3 &1)*16 + (n&15),
// chunk=(km>>4), j=km&7, with km=k&31, kk=k>>5.
static __device__ inline void stage4(_Float16* w, int n, int k,
                                     float4 v) {
  int c = n >> 4, nl = n & 15;
  int kk = k >> 5, km = k & 31;
  int part = (km >> 3) & 3;            // 0..3
  int chunk = part >> 1;
  int lane = (part & 1) * 16 + nl;
  int j = km & 7;                      // 0 or 4 for 4-aligned k
  _Float16* p = w + frag_idx(kk, c, chunk, lane) + j;
  p[0] = (_Float16)v.x; p[1] = (_Float16)v.y;
  p[2] = (_Float16)v.z; p[3] = (_Float16)v.w;
}

// ---------------- gather: h[i,:] = item_emb[x[i],:] ----------------
__global__ __launch_bounds__(256) void gather_k(const int* __restrict__ x,
                                                const float* __restrict__ emb,
                                                float* __restrict__ h) {
  int t = blockIdx.x * 256 + threadIdx.x;
  int node = t >> 5;
  if (node >= NN) return;
  int q = (t & 31) * 4;
  int idx = x[node];
  *(float4*)(h + (size_t)node * DD + q) = *(const float4*)(emb + (size_t)idx * DD + q);
}

// ---------------- scatter: agg[dst] += h[src]; deg[dst] += 1 ----------------
__global__ __launch_bounds__(256) void scatter_k(const int* __restrict__ src,
                                                 const int* __restrict__ dst,
                                                 const float* __restrict__ h,
                                                 float* __restrict__ agg,
                                                 float* __restrict__ deg) {
  int t = blockIdx.x * 256 + threadIdx.x;
  int e = t >> 5;
  if (e >= NE) return;
  int q = (t & 31) * 4;
  int s = src[e], d = dst[e];
  float4 v = *(const float4*)(h + (size_t)s * DD + q);
  float* ap = agg + (size_t)d * DD + q;
  atomicAdd(ap + 0, v.x);
  atomicAdd(ap + 1, v.y);
  atomicAdd(ap + 2, v.z);
  atomicAdd(ap + 3, v.w);
  if (q == 0) atomicAdd(deg + d, 1.0f);
}

// ------- fused SAGE GEMM: out = relu((agg*rdeg)@Wl^T + h@Wr^T + bl) -------
// 8 waves/block, each wave owns a 16-row tile; weights staged in LDS (f16,
// fragment order). A loads software-pipelined over the WMMA chain.
__global__ __launch_bounds__(256) void gemm_dual_k(
    const float* __restrict__ agg, const float* __restrict__ deg,
    const float* __restrict__ h,
    const float* __restrict__ Wl, const float* __restrict__ Wr,
    const float* __restrict__ bias, float* __restrict__ out, int nrows) {
  __shared__ _Float16 sW[2][DD * DD];  // 2 x 32KB, fragment-ordered
  int tid = threadIdx.x;
#pragma unroll 1
  for (int i = tid * 4; i < DD * DD; i += 256 * 4) {
    int n = i >> 7, k = i & 127;
    stage4(&sW[0][0], n, k, *(const float4*)(Wl + i));
    stage4(&sW[1][0], n, k, *(const float4*)(Wr + i));
  }
  __syncthreads();

  int lane = tid & 31;
  int tile = blockIdx.x * 8 + (tid >> 5);
  if (tile * 16 >= nrows) return;  // wave-uniform: EXEC stays all-ones for WMMA
  int m0 = tile * 16;
  int mrow = m0 + (lane & 15);
  const float* aRow = agg + (size_t)mrow * DD;
  const float* hRow = h + (size_t)mrow * DD;
  float rdeg = 1.0f / fmaxf(deg[mrow], 1.0f);

  v8f acc[8];
  v8f zero = {0.f, 0.f, 0.f, 0.f, 0.f, 0.f, 0.f, 0.f};
#pragma unroll
  for (int c = 0; c < 8; ++c) acc[c] = zero;

  // software pipeline: issue next chunk's 8 loads before this chunk's WMMAs
  F32x16 ra = load_row16(aRow, 0, lane);
  F32x16 rh = load_row16(hRow, 0, lane);
#pragma unroll
  for (int kk = 0; kk < 4; ++kk) {
    v16h a1 = cvt_frag(ra, rdeg);   // mean = agg * rdeg, f16
    v16h a2 = cvt_frag(rh, 1.0f);   // h, f16
    if (kk < 3) {
      ra = load_row16(aRow, (kk + 1) * 32, lane);
      rh = load_row16(hRow, (kk + 1) * 32, lane);
    }
#pragma unroll
    for (int c = 0; c < 8; ++c) {
      v16h b1 = b_frag_f(&sW[0][0], kk, c, lane);
      acc[c] = wmma16(a1, b1, acc[c]);
      v16h b2 = b_frag_f(&sW[1][0], kk, c, lane);
      acc[c] = wmma16(a2, b2, acc[c]);
    }
  }

  // epilogue: C/D layout -> M = r + 8*(lane>=16), N = lane&15
  int nlo = lane & 15;
  int mb = m0 + ((lane >> 4) & 1) * 8;
#pragma unroll
  for (int c = 0; c < 8; ++c) {
    float bv = bias[c * 16 + nlo];
#pragma unroll
    for (int r = 0; r < 8; ++r) {
      float v = acc[c][r] + bv;
      out[(size_t)(mb + r) * DD + c * 16 + nlo] = fmaxf(v, 0.0f);
    }
  }
}

// ------------- single GEMM: out = relu(A @ W^T + b) -------------
__global__ __launch_bounds__(256) void gemm_single_k(
    const float* __restrict__ A, const float* __restrict__ W,
    const float* __restrict__ bias, float* __restrict__ out, int nrows) {
  __shared__ _Float16 sW[DD * DD];  // 32KB, fragment-ordered
  int tid = threadIdx.x;
#pragma unroll 1
  for (int i = tid * 4; i < DD * DD; i += 256 * 4) {
    stage4(&sW[0], i >> 7, i & 127, *(const float4*)(W + i));
  }
  __syncthreads();

  int lane = tid & 31;
  int tile = blockIdx.x * 8 + (tid >> 5);
  if (tile * 16 >= nrows) return;
  int m0 = tile * 16;
  const float* aRow = A + (size_t)(m0 + (lane & 15)) * DD;

  v8f acc[8];
  v8f zero = {0.f, 0.f, 0.f, 0.f, 0.f, 0.f, 0.f, 0.f};
#pragma unroll
  for (int c = 0; c < 8; ++c) acc[c] = zero;

  F32x16 ra = load_row16(aRow, 0, lane);
#pragma unroll
  for (int kk = 0; kk < 4; ++kk) {
    v16h a = cvt_frag(ra, 1.0f);
    if (kk < 3) ra = load_row16(aRow, (kk + 1) * 32, lane);
#pragma unroll
    for (int c = 0; c < 8; ++c) {
      v16h b = b_frag_f(&sW[0], kk, c, lane);
      acc[c] = wmma16(a, b, acc[c]);
    }
  }

  int nlo = lane & 15;
  int mb = m0 + ((lane >> 4) & 1) * 8;
#pragma unroll
  for (int c = 0; c < 8; ++c) {
    float bv = bias[c * 16 + nlo];
#pragma unroll
    for (int r = 0; r < 8; ++r) {
      float v = acc[c][r] + bv;
      out[(size_t)(mb + r) * DD + c * 16 + nlo] = fmaxf(v, 0.0f);
    }
  }
}

extern "C" void kernel_launch(void* const* d_in, const int* in_sizes, int n_in,
                              void* d_out, int out_size, void* d_ws, size_t ws_size,
                              hipStream_t stream) {
  (void)in_sizes; (void)n_in; (void)out_size; (void)ws_size;
  const int*   x     = (const int*)d_in[0];
  const int*   ei    = (const int*)d_in[1];
  // d_in[2] = batch (unused)
  const float* item  = (const float*)d_in[3];
  const float* eemb  = (const float*)d_in[4];
  const float* Wl[3] = {(const float*)d_in[5], (const float*)d_in[8],  (const float*)d_in[11]};
  const float* bl[3] = {(const float*)d_in[6], (const float*)d_in[9],  (const float*)d_in[12]};
  const float* Wr[3] = {(const float*)d_in[7], (const float*)d_in[10], (const float*)d_in[13]};
  const float* Wlin1 = (const float*)d_in[14];
  const float* blin1 = (const float*)d_in[15];
  const float* Wlin2 = (const float*)d_in[16];
  const float* blin2 = (const float*)d_in[17];
  float* out = (float*)d_out;

  // workspace carve: hA | hB | agg | deg  (~154 MB)
  float* hA  = (float*)d_ws;
  float* hB  = hA + (size_t)NN * DD;
  float* agg = hB + (size_t)NN * DD;
  float* deg = agg + (size_t)NN * DD;

  const int* src = ei;        // edge_index[0]
  const int* dst = ei + NE;   // edge_index[1]

  gather_k<<<(NN * 32 + 255) / 256, 256, 0, stream>>>(x, item, hA);

  float* cur = hA;
  float* nxt = hB;
  const int gemmNodeBlocks = ((NN / 16) + 7) / 8;   // 782
  const int gemmEdgeBlocks = ((NE / 16) + 7) / 8;   // 12500
  const int scatterBlocks  = (NE * 32) / 256;       // 200000

  for (int l = 0; l < 3; ++l) {
    hipMemsetAsync(agg, 0, (size_t)NN * DD * sizeof(float), stream);
    hipMemsetAsync(deg, 0, (size_t)NN * sizeof(float), stream);
    scatter_k<<<scatterBlocks, 256, 0, stream>>>(src, dst, cur, agg, deg);
    gemm_dual_k<<<gemmNodeBlocks, 256, 0, stream>>>(agg, deg, cur, Wl[l], Wr[l], bl[l], nxt, NN);
    float* t = cur; cur = nxt; nxt = t;
  }

  // h_out = relu(h @ W_lin1^T + b_lin1) -> first NN*DD floats of d_out
  gemm_single_k<<<gemmNodeBlocks, 256, 0, stream>>>(cur, Wlin1, blin1, out, NN);
  // e_out = relu(edge_emb @ W_lin2^T + b_lin2) -> remaining NE*DD floats
  gemm_single_k<<<gemmEdgeBlocks, 256, 0, stream>>>(eemb, Wlin2, blin2, out + (size_t)NN * DD, NE);
}
